// OmegaExpert_26130581029531
// MI455X (gfx1250) — 8-bit matrix operand order repaired and checked by reading, compiled, not run on hardware
//
#include <hip/hip_runtime.h>
#include <cstdint>


#define TOK   16384
#define HID   1024
#define INTER 4096
#define KSTEP 64
#define WEPS  1e-5f

typedef __attribute__((ext_vector_type(8))) int v8i_t;

__device__ __forceinline__ void async_b128(unsigned lds_addr, const void* gaddr) {
  asm volatile("global_load_async_to_lds_b128 %0, %1, off"
               :: "v"(lds_addr), "v"((unsigned long long)(uintptr_t)gaddr)
               : "memory");
}
__device__ __forceinline__ void wait_async_le2() {
  asm volatile("s_wait_asynccnt 0x2" ::: "memory");
}
__device__ __forceinline__ void wait_async_le1() {
  asm volatile("s_wait_asynccnt 0x1" ::: "memory");
}
__device__ __forceinline__ void wait_async_0() {
  asm volatile("s_wait_asynccnt 0x0" ::: "memory");
}

__global__ __launch_bounds__(256) void k_abs_sum_partial(const float* __restrict__ w,
                                                         float* __restrict__ partial) {
  __shared__ float red[256];
  const int t = threadIdx.x;
  const size_t base = (size_t)blockIdx.x * 1024 + (size_t)t * 4;
  float4 v = *(const float4*)(w + base);
  red[t] = fabsf(v.x) + fabsf(v.y) + fabsf(v.z) + fabsf(v.w);
  __syncthreads();
  for (int off = 128; off; off >>= 1) {
    if (t < off) red[t] += red[t + off];
    __syncthreads();
  }
  if (t == 0) partial[blockIdx.x] = red[0];
}

__global__ __launch_bounds__(256) void k_abs_sum_final(const float* __restrict__ partial,
                                                       float* __restrict__ hdr,
                                                       int slot, float inv_numel) {
  __shared__ float red[256];
  const int t = threadIdx.x;
  float s = 0.f;
  for (int i = t; i < 4096; i += 256) s += partial[i];
  red[t] = s;
  __syncthreads();
  for (int off = 128; off; off >>= 1) {
    if (t < off) red[t] += red[t + off];
    __syncthreads();
  }
  if (t == 0) {
    float mean = red[0] * inv_numel;
    float deq  = fmaxf(mean, WEPS);
    hdr[slot]     = 1.0f / deq;
    hdr[3 + slot] = deq;
  }
}

__device__ __forceinline__ signed char tern(float v, float s) {
  return (signed char)(int)fminf(fmaxf(rintf(v * s), -1.f), 1.f);
}

__global__ __launch_bounds__(256) void k_weight_quant(const float* __restrict__ w,
                                                      int8_t* __restrict__ q,
                                                      const float* __restrict__ hdr, int slot) {
  const float s = hdr[slot];
  const size_t base = (size_t)blockIdx.x * 1024 + (size_t)threadIdx.x * 4;
  float4 v = *(const float4*)(w + base);
  char4 c;
  c.x = tern(v.x, s); c.y = tern(v.y, s); c.z = tern(v.z, s); c.w = tern(v.w, s);
  *(char4*)(q + base) = c;
}

__device__ __forceinline__ int pack4(float4 v, float s) {
  int b0 = (int)fminf(fmaxf(rintf(v.x * s), -128.f), 127.f) & 255;
  int b1 = (int)fminf(fmaxf(rintf(v.y * s), -128.f), 127.f) & 255;
  int b2 = (int)fminf(fmaxf(rintf(v.z * s), -128.f), 127.f) & 255;
  int b3 = (int)fminf(fmaxf(rintf(v.w * s), -128.f), 127.f) & 255;
  return b0 | (b1 << 8) | (b2 << 16) | (b3 << 24);
}

__global__ __launch_bounds__(256) void k_act_quant_x(const float* __restrict__ x,
                                                     int8_t* __restrict__ q,
                                                     float* __restrict__ deq) {
  __shared__ float red[256];
  const int t = threadIdx.x;
  const size_t row = (size_t)blockIdx.x * HID;
  float4 v = *(const float4*)(x + row + (size_t)t * 4);
  red[t] = fmaxf(fmaxf(fabsf(v.x), fabsf(v.y)), fmaxf(fabsf(v.z), fabsf(v.w)));
  __syncthreads();
  for (int off = 128; off; off >>= 1) {
    if (t < off) red[t] = fmaxf(red[t], red[t + off]);
    __syncthreads();
  }
  const float mx = fmaxf(red[0], WEPS);
  const float scale = 127.f / mx;
  if (t == 0) deq[blockIdx.x] = mx * (1.f / 127.f);
  *(int*)(q + row + (size_t)t * 4) = pack4(v, scale);
}

__global__ __launch_bounds__(256) void k_act_quant_h(const float* __restrict__ h,
                                                     int8_t* __restrict__ q,
                                                     float* __restrict__ deq) {
  __shared__ float red[256];
  const int t = threadIdx.x;
  const size_t row = (size_t)blockIdx.x * INTER;
  const float* p = h + row + (size_t)t * 16;
  float4 v0 = *(const float4*)(p + 0);
  float4 v1 = *(const float4*)(p + 4);
  float4 v2 = *(const float4*)(p + 8);
  float4 v3 = *(const float4*)(p + 12);
  float m = 0.f;
  m = fmaxf(m, fmaxf(fmaxf(fabsf(v0.x), fabsf(v0.y)), fmaxf(fabsf(v0.z), fabsf(v0.w))));
  m = fmaxf(m, fmaxf(fmaxf(fabsf(v1.x), fabsf(v1.y)), fmaxf(fabsf(v1.z), fabsf(v1.w))));
  m = fmaxf(m, fmaxf(fmaxf(fabsf(v2.x), fabsf(v2.y)), fmaxf(fabsf(v2.z), fabsf(v2.w))));
  m = fmaxf(m, fmaxf(fmaxf(fabsf(v3.x), fabsf(v3.y)), fmaxf(fabsf(v3.z), fabsf(v3.w))));
  red[t] = m;
  __syncthreads();
  for (int off = 128; off; off >>= 1) {
    if (t < off) red[t] = fmaxf(red[t], red[t + off]);
    __syncthreads();
  }
  const float mx = fmaxf(red[0], WEPS);
  const float scale = 127.f / mx;
  if (t == 0) deq[blockIdx.x] = mx * (1.f / 127.f);
  int4 pk;
  pk.x = pack4(v0, scale); pk.y = pack4(v1, scale);
  pk.z = pack4(v2, scale); pk.w = pack4(v3, scale);
  *(int4*)(q + row + (size_t)t * 16) = pk;
}

__device__ __forceinline__ v8i_t load_a64(const int8_t* __restrict__ arow, int kh) {
  v8i_t a = {0, 0, 0, 0, 0, 0, 0, 0};
#pragma unroll
  for (int v = 0; v < 8; ++v) {
    const int koff = ((v >> 2) << 5) + (kh << 4) + ((v & 3) << 2);
    a[v] = *(const int*)(arow + koff);
  }
  return a;
}
__device__ __forceinline__ v8i_t load_b_lds(const int8_t* __restrict__ tile, int brow, int kh) {
  const int8_t* p = tile + brow * KSTEP + kh * 16;
  int4 b0 = *(const int4*)(p);
  int4 b1 = *(const int4*)(p + 32);
  v8i_t b = {b0.x, b0.y, b0.z, b0.w, b1.x, b1.y, b1.z, b1.w};
  return b;
}

__global__ __launch_bounds__(256) void k_gemm_gateup(const int8_t* __restrict__ qx,
                                                     const int8_t* __restrict__ qg,
                                                     const int8_t* __restrict__ qu,
                                                     const float* __restrict__ deqx,
                                                     const float* __restrict__ hdr,
                                                     float* __restrict__ h) {
  __shared__ int8_t lds_g[2][64 * KSTEP];
  __shared__ int8_t lds_u[2][64 * KSTEP];
  const int t = threadIdx.x;
  const int lane = t & 31;
  const int wave = t >> 5;
  const int kh = lane >> 4;
  const int ln = lane & 15;
  const int m0 = (blockIdx.x * 8 + wave) * 16;
  const int n0 = blockIdx.y * 64;
  const float dg = hdr[3];
  const float du = hdr[4];

  const int crow  = t >> 2;
  const int ckoff = (t & 3) * 16;
  const int8_t* gsrc_g = qg + (size_t)(n0 + crow) * HID + ckoff;
  const int8_t* gsrc_u = qu + (size_t)(n0 + crow) * HID + ckoff;
  const int ldst = crow * KSTEP + ckoff;

  const v8i_t zero = {0, 0, 0, 0, 0, 0, 0, 0};
  v8i_t accG[4] = {zero, zero, zero, zero};
  v8i_t accU[4] = {zero, zero, zero, zero};

  const int8_t* arow = qx + (size_t)(m0 + ln) * HID;
  constexpr int S = HID / KSTEP;

  async_b128((unsigned)(size_t)&lds_g[0][ldst], gsrc_g);
  async_b128((unsigned)(size_t)&lds_u[0][ldst], gsrc_u);
  async_b128((unsigned)(size_t)&lds_g[1][ldst], gsrc_g + KSTEP);
  async_b128((unsigned)(size_t)&lds_u[1][ldst], gsrc_u + KSTEP);

  for (int i = 0; i < S; ++i) {
    if (i + 1 < S) wait_async_le2();
    else           wait_async_0();
    __syncthreads();

    const int p = i & 1;
    const v8i_t a = load_a64(arow + i * KSTEP, kh);
#pragma unroll
    for (int j = 0; j < 4; ++j) {
      const int brow = j * 16 + ln;
      const v8i_t vbg = load_b_lds(lds_g[p], brow, kh);
      const v8i_t vbu = load_b_lds(lds_u[p], brow, kh);
      accG[j] = __builtin_amdgcn_wmma_i32_16x16x64_iu8(true, a, true, vbg, accG[j], false, false);
      accU[j] = __builtin_amdgcn_wmma_i32_16x16x64_iu8(true, a, true, vbu, accU[j], false, false);
    }
    __syncthreads();
    if (i + 2 < S) {
      const int koff = (i + 2) * KSTEP;
      async_b128((unsigned)(size_t)&lds_g[p][ldst], gsrc_g + koff);
      async_b128((unsigned)(size_t)&lds_u[p][ldst], gsrc_u + koff);
    }
  }

  float ax[8];
#pragma unroll
  for (int r = 0; r < 8; ++r) ax[r] = deqx[m0 + r + 8 * kh];
#pragma unroll
  for (int j = 0; j < 4; ++j) {
#pragma unroll
    for (int r = 0; r < 8; ++r) {
      const int token = m0 + r + 8 * kh;
      const float g = (float)accG[j][r] * (ax[r] * dg);
      const float u = (float)accU[j][r] * (ax[r] * du);
      const float rg = g > 0.f ? g : 0.f;
      h[(size_t)token * INTER + n0 + j * 16 + ln] = rg * rg * u;
    }
  }
}

__global__ __launch_bounds__(256) void k_gemm_down(const int8_t* __restrict__ qh,
                                                   const int8_t* __restrict__ qd,
                                                   const float* __restrict__ deqh,
                                                   const float* __restrict__ hdr,
                                                   float* __restrict__ out) {
  __shared__ int8_t lds_d[2][64 * KSTEP];
  const int t = threadIdx.x;
  const int lane = t & 31;
  const int wave = t >> 5;
  const int kh = lane >> 4;
  const int ln = lane & 15;
  const int m0 = (blockIdx.x * 8 + wave) * 16;
  const int n0 = blockIdx.y * 64;
  const float dd = hdr[5];

  const int crow  = t >> 2;
  const int ckoff = (t & 3) * 16;
  const int8_t* gsrc = qd + (size_t)(n0 + crow) * INTER + ckoff;
  const int ldst = crow * KSTEP + ckoff;

  const v8i_t zero = {0, 0, 0, 0, 0, 0, 0, 0};
  v8i_t acc[4] = {zero, zero, zero, zero};

  const int8_t* arow = qh + (size_t)(m0 + ln) * INTER;
  constexpr int S = INTER / KSTEP;

  async_b128((unsigned)(size_t)&lds_d[0][ldst], gsrc);
  async_b128((unsigned)(size_t)&lds_d[1][ldst], gsrc + KSTEP);

  for (int i = 0; i < S; ++i) {
    if (i + 1 < S) wait_async_le1();
    else           wait_async_0();
    __syncthreads();

    const int p = i & 1;
    const v8i_t a = load_a64(arow + i * KSTEP, kh);
#pragma unroll
    for (int j = 0; j < 4; ++j) {
      const v8i_t b = load_b_lds(lds_d[p], j * 16 + ln, kh);
      acc[j] = __builtin_amdgcn_wmma_i32_16x16x64_iu8(true, a, true, b, acc[j], false, false);
    }
    __syncthreads();
    if (i + 2 < S) async_b128((unsigned)(size_t)&lds_d[p][ldst], gsrc + (i + 2) * KSTEP);
  }

  float ah[8];
#pragma unroll
  for (int r = 0; r < 8; ++r) ah[r] = deqh[m0 + r + 8 * kh];
#pragma unroll
  for (int j = 0; j < 4; ++j) {
#pragma unroll
    for (int r = 0; r < 8; ++r) {
      const int token = m0 + r + 8 * kh;
      out[(size_t)token * HID + n0 + j * 16 + ln] = (float)acc[j][r] * (ah[r] * dd);
    }
  }
}

extern "C" void kernel_launch(void* const* d_in, const int* in_sizes, int n_in,
                              void* d_out, int out_size, void* d_ws, size_t ws_size,
                              hipStream_t stream) {
  const float* x      = (const float*)d_in[0];
  const float* w_gate = (const float*)d_in[1];
  const float* w_up   = (const float*)d_in[2];
  const float* w_down = (const float*)d_in[3];
  float* out = (float*)d_out;

  char* ws = (char*)d_ws;
  float*  hdr     = (float*)ws;
  float*  partial = (float*)(ws + 256);
  int8_t* qg      = (int8_t*)(ws + 256 + 16384);
  int8_t* qu      = qg + (size_t)INTER * HID;
  int8_t* qd      = qu + (size_t)INTER * HID;
  int8_t* qx      = qd + (size_t)HID * INTER;
  float*  deqx    = (float*)(qx + (size_t)TOK * HID);
  float*  deqh    = deqx + TOK;
  int8_t* qh      = (int8_t*)(deqh + TOK);
  float*  hbuf    = (float*)(qh + (size_t)TOK * INTER);

  const float inv_numel = 1.0f / (float)((size_t)INTER * HID);

  k_abs_sum_partial<<<4096, 256, 0, stream>>>(w_gate, partial);
  k_abs_sum_final  <<<1, 256, 0, stream>>>(partial, hdr, 0, inv_numel);
  k_abs_sum_partial<<<4096, 256, 0, stream>>>(w_up, partial);
  k_abs_sum_final  <<<1, 256, 0, stream>>>(partial, hdr, 1, inv_numel);
  k_abs_sum_partial<<<4096, 256, 0, stream>>>(w_down, partial);
  k_abs_sum_final  <<<1, 256, 0, stream>>>(partial, hdr, 2, inv_numel);

  k_weight_quant<<<4096, 256, 0, stream>>>(w_gate, qg, hdr, 0);
  k_weight_quant<<<4096, 256, 0, stream>>>(w_up,   qu, hdr, 1);
  k_weight_quant<<<4096, 256, 0, stream>>>(w_down, qd, hdr, 2);

  k_act_quant_x<<<TOK, 256, 0, stream>>>(x, qx, deqx);

  k_gemm_gateup<<<dim3(TOK / 128, INTER / 64), 256, 0, stream>>>(qx, qg, qu, deqx, hdr, hbuf);

  k_act_quant_h<<<TOK, 256, 0, stream>>>(hbuf, qh, deqh);

  k_gemm_down<<<dim3(TOK / 128, HID / 64), 256, 0, stream>>>(qh, qd, deqh, hdr, out);
}
